// ScalableSuperMultiHeadAttention_50405736186197
// MI455X (gfx1250) — matrix operand order repaired and checked by reading, not compiled, not run on hardware
//
#include <hip/hip_runtime.h>
#include <hip/hip_bf16.h>
#include <math.h>

typedef _Float16 half_t;
typedef __attribute__((ext_vector_type(16))) _Float16 v16h;
typedef __attribute__((ext_vector_type(8)))  _Float16 v8h;
typedef __attribute__((ext_vector_type(8)))  float    v8f;

static constexpr int Bb = 2, Nn = 2048, Cc = 768, Hh = 24, HD = 32;

#define WMMA_F16(a, b, c) \
    __builtin_amdgcn_wmma_f32_16x16x32_f16(false, (a), false, (b), (short)0, (c), false, false)

#define CAT16(lo, hi) \
    __builtin_shufflevector((lo), (hi), 0, 1, 2, 3, 4, 5, 6, 7, 8, 9, 10, 11, 12, 13, 14, 15)

// ---------------------------------------------------------------------------
// Fragment helpers (wave32 layouts per CDNA5 ISA 7.12.2)
// ---------------------------------------------------------------------------
__device__ __forceinline__ v8f vzero8() {
    v8f z;
#pragma unroll
    for (int i = 0; i < 8; i++) z[i] = 0.f;
    return z;
}

// A (16x32 f16, MxK), row-major source, ld in halfs.
// lane<16: row=lane, K 0..7 & 16..23 ; lane>=16: row=lane-16, K 8..15 & 24..31
__device__ __forceinline__ v16h load_A_frag(const half_t* __restrict__ p, int ld, int lane) {
    int r = lane & 15, hf = lane >> 4;
    const half_t* row = p + (size_t)r * ld;
    v8h lo = *(const v8h*)(row + hf * 8);
    v8h hi = *(const v8h*)(row + 16 + hf * 8);
    return CAT16(lo, hi);
}

// B (32x16 f16, KxN) where each column's K values are CONTIGUOUS: B(k,n)=p[n*ld+k].
// lane<16: col=lane, K 0..7 & 16..23 ; lane>=16: col=lane-16, K 8..15 & 24..31 (two b128 loads).
__device__ __forceinline__ v16h load_B_kc(const half_t* __restrict__ p, int ld, int lane) {
    int n = lane & 15, kb = (lane >> 4) * 8;
    const half_t* col = p + (size_t)n * ld;
    v8h lo = *(const v8h*)(col + kb);
    v8h hi = *(const v8h*)(col + 16 + kb);
    return CAT16(lo, hi);
}

// ---------------------------------------------------------------------------
// Elementwise cast f32 -> f16
// ---------------------------------------------------------------------------
__global__ void cast_f16_kernel(const float* __restrict__ a, half_t* __restrict__ o, int n) {
    int i = blockIdx.x * 256 + threadIdx.x;
    if (i < n) o[i] = (half_t)a[i];
}

// ---------------------------------------------------------------------------
// RMSNorm (+ cast to f16). One block per row of C=768.
// WRT: additionally scatter the per-head TRANSPOSED copy (HD x N per (b,h)),
// which feeds the premix GEMM with fully contiguous A fragments.
// Head-view mapping: flat = n0*C + c = h*(N*HD) + n'*HD + d, all powers of two.
// ---------------------------------------------------------------------------
template <bool WRT>
__global__ void __launch_bounds__(256)
rmsnorm_cast_kernel(const float* __restrict__ xin, const float* __restrict__ g,
                    half_t* __restrict__ outh, half_t* __restrict__ outT) {
    int row = blockIdx.x;  // b*Nn + n0
    const float* xr = xin + (size_t)row * Cc;
    float ss = 0.f;
    for (int c = threadIdx.x; c < Cc; c += 256) { float v = xr[c]; ss += v * v; }
#pragma unroll
    for (int m = 1; m < 32; m <<= 1) ss += __shfl_xor(ss, m, 32);
    __shared__ float red[8];
    int w = threadIdx.x >> 5, lane = threadIdx.x & 31;
    if (lane == 0) red[w] = ss;
    __syncthreads();
    float tot = 0.f;
#pragma unroll
    for (int i = 0; i < 8; i++) tot += red[i];
    float r = rsqrtf(tot / (float)Cc + 1.1920929e-07f);  // finfo(f32).eps
    half_t* orow = outh + (size_t)row * Cc;
    int b = row >> 11;          // row / Nn
    int n0 = row & (Nn - 1);    // row % Nn
    for (int c = threadIdx.x; c < Cc; c += 256) {
        half_t val = (half_t)(xr[c] * r * g[c]);
        orow[c] = val;
        if (WRT) {
            size_t flat = (size_t)n0 * Cc + c;   // within batch b
            int h  = (int)(flat >> 16);          // / (Nn*HD)
            int np = (int)((flat >> 5) & (Nn - 1));
            int d  = (int)(flat & (HD - 1));
            outT[(((size_t)(b * Hh + h) * HD + d) << 11) + np] = val;
        }
    }
}

// ---------------------------------------------------------------------------
// GEMM y = A @ W^T ; A (M x K) f16 row-major, W (Nc x K) f16 row-major.
// Block = 8 waves (4x2), block tile 128x64, wave tile 32x32 (2x2 WMMA frags).
// EPI: 0 = store f16 | 1 = exact GELU, store f16 | 2 = add residual, store f32
// ---------------------------------------------------------------------------
template <int EPI>
__global__ void __launch_bounds__(256)
gemm_nt_kernel(const half_t* __restrict__ A, const half_t* __restrict__ W,
               int M, int Nc, int K,
               half_t* __restrict__ outh, float* __restrict__ outf,
               const float* __restrict__ resid) {
    int lane = threadIdx.x & 31;
    int w = threadIdx.x >> 5;
    int wm = w & 3, wn = w >> 2;
    int rb = blockIdx.x * 128 + wm * 32;
    int cb = blockIdx.y * 64 + wn * 32;

    v8f acc[2][2];
#pragma unroll
    for (int i = 0; i < 2; i++)
#pragma unroll
        for (int j = 0; j < 2; j++) acc[i][j] = vzero8();

    for (int k = 0; k < K; k += 32) {
        v16h a0 = load_A_frag(A + (size_t)rb * K + k, K, lane);
        v16h a1 = load_A_frag(A + (size_t)(rb + 16) * K + k, K, lane);
        v16h b0 = load_B_kc(W + (size_t)cb * K + k, K, lane);
        v16h b1 = load_B_kc(W + (size_t)(cb + 16) * K + k, K, lane);
        acc[0][0] = WMMA_F16(a0, b0, acc[0][0]);
        acc[0][1] = WMMA_F16(a0, b1, acc[0][1]);
        acc[1][0] = WMMA_F16(a1, b0, acc[1][0]);
        acc[1][1] = WMMA_F16(a1, b1, acc[1][1]);
    }

    int hf = lane >> 4, cn = lane & 15;
#pragma unroll
    for (int ri = 0; ri < 2; ri++)
#pragma unroll
        for (int ci = 0; ci < 2; ci++)
#pragma unroll
            for (int i = 0; i < 8; i++) {
                int r = rb + ri * 16 + i + 8 * hf;
                int c = cb + ci * 16 + cn;
                float v = acc[ri][ci][i];
                size_t idx = (size_t)r * Nc + c;
                if (EPI == 0) {
                    outh[idx] = (half_t)v;
                } else if (EPI == 1) {
                    float gl = 0.5f * v * (1.f + erff(v * 0.70710678118654752f));
                    outh[idx] = (half_t)gl;
                } else {
                    outf[idx] = v + resid[idx];
                }
            }
}

// ---------------------------------------------------------------------------
// Value premix, computed TRANSPOSED: v'^T (HD x N) = v^T (HD x N) @ WA^T.
//   v'^T[d,m] = sum_k v^T[d,k] * WA[m,k]
// A = xnT per-(b,h) slab (rows contiguous), B = WA rows (K-contiguous).
// Block = 8 waves, each wave 32(dims) x 32(dest keys); grid (N/256, B*H).
// Output v'^T lands directly in the layout the attention V fragments want.
// ---------------------------------------------------------------------------
__global__ void __launch_bounds__(256)
premix_kernel(const half_t* __restrict__ WA, const half_t* __restrict__ xnT,
              half_t* __restrict__ vT) {
    int lane = threadIdx.x & 31, w = threadIdx.x >> 5;
    int bh = blockIdx.y;
    const half_t* aT = xnT + (size_t)bh * HD * Nn;  // (32 x 2048) row-major
    half_t* vdst = vT + (size_t)bh * HD * Nn;       // (32 x 2048) row-major
    int cbase = blockIdx.x * 256 + w * 32;          // destination keys

    v8f acc[2][2];
#pragma unroll
    for (int i = 0; i < 2; i++)
#pragma unroll
        for (int j = 0; j < 2; j++) acc[i][j] = vzero8();

    for (int k = 0; k < Nn; k += 32) {
        v16h a0 = load_A_frag(aT + k, Nn, lane);                   // dims 0..15
        v16h a1 = load_A_frag(aT + (size_t)16 * Nn + k, Nn, lane); // dims 16..31
        v16h b0 = load_B_kc(WA + (size_t)cbase * Nn + k, Nn, lane);
        v16h b1 = load_B_kc(WA + (size_t)(cbase + 16) * Nn + k, Nn, lane);
        acc[0][0] = WMMA_F16(a0, b0, acc[0][0]);
        acc[0][1] = WMMA_F16(a0, b1, acc[0][1]);
        acc[1][0] = WMMA_F16(a1, b0, acc[1][0]);
        acc[1][1] = WMMA_F16(a1, b1, acc[1][1]);
    }

    int hf = lane >> 4, cn = lane & 15;
#pragma unroll
    for (int ri = 0; ri < 2; ri++)
#pragma unroll
        for (int ci = 0; ci < 2; ci++)
#pragma unroll
            for (int i = 0; i < 8; i++) {
                int d = ri * 16 + i + 8 * hf;       // output dim (C-layout row)
                int m = cbase + ci * 16 + cn;       // dest key (C-layout col)
                vdst[((size_t)d << 11) + m] = (half_t)acc[ri][ci][i];
            }
}

// ---------------------------------------------------------------------------
// Flash-style attention per (b,h): attn = softmax(coef * q k^T) @ v' + x.
// Each wave owns 16 query rows; streams 32-key blocks with online softmax.
// HD=32 == one WMMA K-step. coef is pre-folded into the q fragment.
// P goes through per-wave LDS for C-layout -> A-layout conversion.
// ---------------------------------------------------------------------------
__global__ void __launch_bounds__(256)
attn_kernel(const half_t* __restrict__ q16, const half_t* __restrict__ xn,
            const half_t* __restrict__ vT, const float* __restrict__ scale_param,
            const float* __restrict__ x, float* __restrict__ attnres) {
    __shared__ __align__(16) half_t Plds[8][16 * 40];  // 16x32 P tile per wave, padded
    int lane = threadIdx.x & 31, w = threadIdx.x >> 5;
    int bh = blockIdx.y, b = bh / Hh, h = bh % Hh;
    size_t base = (size_t)b * Nn * Cc + (size_t)h * Nn * HD;
    const half_t* qh = q16 + base;                 // (N x 32)
    const half_t* kh = xn + base;                  // (N x 32)
    const half_t* vh = vT + (size_t)bh * HD * Nn;  // (32 x N)
    float coef = logf((float)Cc) * scale_param[h] * 5.65685424949238f;  // log(C)*s*sqrt(HD)

    int rb = blockIdx.x * 128 + w * 16;
    v16h aq = load_A_frag(qh + (size_t)rb * HD, HD, lane);
#pragma unroll
    for (int j = 0; j < 16; j++) aq[j] = (half_t)((float)aq[j] * coef);

    float mr[8], lr[8];
    v8f acc0 = vzero8(), acc1 = vzero8();
#pragma unroll
    for (int i = 0; i < 8; i++) { mr[i] = -1e30f; lr[i] = 0.f; }

    int hf = lane >> 4, cn = lane & 15;
    half_t* P = Plds[w];

    for (int kb = 0; kb < Nn; kb += 32) {
        // S = q (16x32) @ k^T : k rows are the B columns, K(=HD) contiguous
        v16h bk0 = load_B_kc(kh + (size_t)kb * HD, HD, lane);
        v16h bk1 = load_B_kc(kh + (size_t)(kb + 16) * HD, HD, lane);
        v8f z = vzero8();
        v8f s0 = WMMA_F16(aq, bk0, z);
        v8f s1 = WMMA_F16(aq, bk1, z);

        // Online softmax (rows live across the 16 lanes of each half-wave)
#pragma unroll
        for (int i = 0; i < 8; i++) {
            float t = fmaxf(s0[i], s1[i]);
#pragma unroll
            for (int m = 1; m < 16; m <<= 1) t = fmaxf(t, __shfl_xor(t, m, 32));
            float mnew = fmaxf(mr[i], t);
            float corr = expf(mr[i] - mnew);
            float p0 = expf(s0[i] - mnew);
            float p1 = expf(s1[i] - mnew);
            float rs = p0 + p1;
#pragma unroll
            for (int m = 1; m < 16; m <<= 1) rs += __shfl_xor(rs, m, 32);
            lr[i] = lr[i] * corr + rs;
            mr[i] = mnew;
            acc0[i] *= corr;
            acc1[i] *= corr;
            int r = i + 8 * hf;  // C-layout row
            P[r * 40 + cn]      = (half_t)p0;
            P[r * 40 + 16 + cn] = (half_t)p1;
        }

        // DS ops are in-order within a wave; just drain the DS counter.
        __asm__ volatile("s_wait_dscnt 0" ::: "memory");

        // Re-read P in the A-matrix 16x32 layout (two aligned v8h, concat)
        const half_t* prow = P + cn * 40;
        v8h plo = *(const v8h*)(prow + hf * 8);
        v8h phi = *(const v8h*)(prow + 16 + hf * 8);
        v16h pa = CAT16(plo, phi);

        // V fragments: vT is (dim x key), so K (= key) is contiguous per column
        v16h bv0 = load_B_kc(vh + kb, Nn, lane);                      // dims 0..15
        v16h bv1 = load_B_kc(vh + (size_t)16 * Nn + kb, Nn, lane);    // dims 16..31
        acc0 = WMMA_F16(pa, bv0, acc0);
        acc1 = WMMA_F16(pa, bv1, acc1);
    }

    // Normalize + residual, scatter back into the "reshaped" flat layout
#pragma unroll
    for (int i = 0; i < 8; i++) {
        float inv = 1.f / lr[i];
        int r = rb + i + 8 * hf;
        size_t i0 = base + (size_t)r * HD + cn;
        size_t i1 = i0 + 16;
        attnres[i0] = acc0[i] * inv + x[i0];
        attnres[i1] = acc1[i] * inv + x[i1];
    }
}

// ---------------------------------------------------------------------------
// Orchestration
// ---------------------------------------------------------------------------
extern "C" void kernel_launch(void* const* d_in, const int* in_sizes, int n_in,
                              void* d_out, int out_size, void* d_ws, size_t ws_size,
                              hipStream_t stream) {
    const float* x  = (const float*)d_in[0];
    const float* sp = (const float*)d_in[1];
    const float* Wq = (const float*)d_in[2];
    const float* WA = (const float*)d_in[3];
    const float* W1 = (const float*)d_in[4];
    const float* W2 = (const float*)d_in[5];
    const float* g1 = (const float*)d_in[6];
    const float* g2 = (const float*)d_in[7];
    float* out = (float*)d_out;
    (void)in_sizes; (void)n_in; (void)out_size; (void)ws_size;

    char* ws = (char*)d_ws;
    size_t off = 0;
    auto take = [&](size_t bytes) -> void* {
        void* p = ws + off;
        off += (bytes + 255) & ~(size_t)255;
        return p;
    };
    half_t* xn16 = (half_t*)take((size_t)Bb * Nn * Cc * 2);
    half_t* xnT  = (half_t*)take((size_t)Bb * Hh * HD * Nn * 2);
    half_t* q16  = (half_t*)take((size_t)Bb * Nn * Cc * 2);
    half_t* wq16 = (half_t*)take((size_t)Cc * Cc * 2);
    half_t* wa16 = (half_t*)take((size_t)Nn * Nn * 2);
    half_t* w116 = (half_t*)take((size_t)Cc * Cc * 2);
    half_t* w216 = (half_t*)take((size_t)Cc * Cc * 2);
    half_t* vT16 = (half_t*)take((size_t)Bb * Hh * HD * Nn * 2);
    float*  ares = (float*)take((size_t)Bb * Nn * Cc * 4);
    half_t* h16  = (half_t*)take((size_t)Bb * Nn * Cc * 2);
    half_t* t16  = (half_t*)take((size_t)Bb * Nn * Cc * 2);

    // Weight casts (f32 -> f16), all in L2-resident workspace
    cast_f16_kernel<<<(Cc * Cc + 255) / 256, 256, 0, stream>>>(Wq, wq16, Cc * Cc);
    cast_f16_kernel<<<(Nn * Nn + 255) / 256, 256, 0, stream>>>(WA, wa16, Nn * Nn);
    cast_f16_kernel<<<(Cc * Cc + 255) / 256, 256, 0, stream>>>(W1, w116, Cc * Cc);
    cast_f16_kernel<<<(Cc * Cc + 255) / 256, 256, 0, stream>>>(W2, w216, Cc * Cc);

    // 1) RMSNorm1 + cast, also emitting the per-head transposed slab for premix
    rmsnorm_cast_kernel<true><<<Bb * Nn, 256, 0, stream>>>(x, g1, xn16, xnT);

    // 2) q = xn @ Wq^T  (4096 x 768 x 768)
    gemm_nt_kernel<0><<<dim3(32, 12), 256, 0, stream>>>(xn16, wq16, Bb * Nn, Cc, Cc,
                                                        q16, nullptr, nullptr);

    // 3) value premix per (b,h): v'^T = v^T @ WA^T (all-contiguous fragments)
    premix_kernel<<<dim3(Nn / 256, Bb * Hh), 256, 0, stream>>>(wa16, xnT, vT16);

    // 4) flash attention + residual
    attn_kernel<<<dim3(Nn / 128, Bb * Hh), 256, 0, stream>>>(q16, xn16, vT16, sp, x, ares);

    // 5) RMSNorm2 + cast (no transposed copy needed)
    rmsnorm_cast_kernel<false><<<Bb * Nn, 256, 0, stream>>>(ares, g2, h16, nullptr);

    // 6) FF1 with exact-GELU epilogue
    gemm_nt_kernel<1><<<dim3(32, 12), 256, 0, stream>>>(h16, w116, Bb * Nn, Cc, Cc,
                                                        t16, nullptr, nullptr);

    // 7) FF2 with residual-add epilogue -> f32 output
    gemm_nt_kernel<2><<<dim3(32, 12), 256, 0, stream>>>(t16, w216, Bb * Nn, Cc, Cc,
                                                        nullptr, out, ares);
}
